// MT_Module_new_25168508354943
// MI455X (gfx1250) — compile-verified, matrix operand order checked by reading, not run on hardware
//
#include <hip/hip_runtime.h>
#include <hip/hip_bf16.h>
#include <math.h>

typedef __attribute__((ext_vector_type(16))) _Float16 v16h;
typedef __attribute__((ext_vector_type(8)))  _Float16 v8h;
typedef __attribute__((ext_vector_type(8)))  float    v8f;

#define QK_CH 64
#define QC    16
#define VC    128
#define NPOS  4096
#define BATCH 4
#define KCH   64              // keys per chunk
#define NCH   (NPOS / KCH)    // 64 chunks

// ---------------------------------------------------------------------------
// gfx1250 async global->LDS copy (ASYNCcnt path), 16B per lane.
// Issued via inline asm on purpose: the clang builtin is modeled as an
// LDS-writing memory op, which makes __syncthreads() drain ALL outstanding
// loads (s_wait_loadcnt 0) and kills the double-buffer overlap. With opaque
// asm, completion is handshaken explicitly with s_wait_asynccnt.
// ---------------------------------------------------------------------------
__device__ __forceinline__ void async_ld16(const _Float16* g, _Float16* l) {
    unsigned loff = (unsigned)(size_t)(void*)l;   // low 32 bits = LDS offset
    void* gv = (void*)g;
    asm volatile("global_load_async_to_lds_b128 %0, %1, off"
                 :: "v"(loff), "v"(gv) : "memory");
}

#define WAIT_ASYNC(N) asm volatile("s_wait_asynccnt %0" :: "n"(N) : "memory")

// compiler-level LDS ordering fence (DS ops are in-order per wave in HW)
#define LDS_FENCE() do { __builtin_amdgcn_wave_barrier(); \
                         asm volatile("" ::: "memory"); } while (0)

// ---------------------------------------------------------------------------
// 16-lane-group xor butterfly max via v_permlane16_b32 (VALU, no LDS)
// ---------------------------------------------------------------------------
#if __has_builtin(__builtin_amdgcn_permlane16)
__device__ __forceinline__ float permxor(float v, unsigned lo, unsigned hi) {
    int s = __float_as_int(v);
    int r = __builtin_amdgcn_permlane16(s, s, lo, hi, false, false);
    return __int_as_float(r);
}
__device__ __forceinline__ float xor16_max(float v) {
    v = fmaxf(v, permxor(v, 0x67452301u, 0xEFCDAB89u));  // xor 1
    v = fmaxf(v, permxor(v, 0x54761032u, 0xDCFE98BAu));  // xor 2
    v = fmaxf(v, permxor(v, 0x32107654u, 0xBA98FEDCu));  // xor 4
    v = fmaxf(v, permxor(v, 0xFEDCBA98u, 0x76543210u));  // xor 8
    return v;
}
#else
__device__ __forceinline__ float xor16_max(float v) {
    v = fmaxf(v, __shfl_xor(v, 1, 32));
    v = fmaxf(v, __shfl_xor(v, 2, 32));
    v = fmaxf(v, __shfl_xor(v, 4, 32));
    v = fmaxf(v, __shfl_xor(v, 8, 32));
    return v;
}
#endif

// ---------------------------------------------------------------------------
// Kernel 1: q,k projections (64 -> 16 ch), f16 [B][pos][32], zero-padded K
// ---------------------------------------------------------------------------
__global__ void __launch_bounds__(64)
proj_qk_kernel(const float* __restrict__ x,
               const float* __restrict__ Wq, const float* __restrict__ bq,
               const float* __restrict__ Wk, const float* __restrict__ bk,
               _Float16* __restrict__ qh, _Float16* __restrict__ kh)
{
    __shared__ float sWq[QC * QK_CH];
    __shared__ float sWk[QC * QK_CH];
    __shared__ float sbq[QC], sbk[QC];
    const int tx = threadIdx.x;
    for (int idx = tx; idx < QC * QK_CH; idx += 64) {
        sWq[idx] = Wq[idx];
        sWk[idx] = Wk[idx];
    }
    if (tx < QC) { sbq[tx] = bq[tx]; sbk[tx] = bk[tx]; }
    __syncthreads();

    const int b  = blockIdx.x / (NPOS / 64);
    const int it = blockIdx.x % (NPOS / 64);
    const int i  = it * 64 + tx;

    float aq[QC], ak[QC];
#pragma unroll
    for (int d = 0; d < QC; ++d) { aq[d] = sbq[d]; ak[d] = sbk[d]; }

    const float* xp = x + (size_t)b * QK_CH * NPOS + i;
    for (int c = 0; c < QK_CH; ++c) {
        float xv = xp[(size_t)c * NPOS];
#pragma unroll
        for (int d = 0; d < QC; ++d) {
            aq[d] = fmaf(sWq[d * QK_CH + c], xv, aq[d]);
            ak[d] = fmaf(sWk[d * QK_CH + c], xv, ak[d]);
        }
    }
    _Float16* qo = qh + ((size_t)b * NPOS + i) * 32;
    _Float16* ko = kh + ((size_t)b * NPOS + i) * 32;
#pragma unroll
    for (int d = 0; d < QC; ++d) { qo[d] = (_Float16)aq[d]; ko[d] = (_Float16)ak[d]; }
#pragma unroll
    for (int d = QC; d < 32; ++d) { qo[d] = (_Float16)0.f; ko[d] = (_Float16)0.f; }
}

// ---------------------------------------------------------------------------
// Kernel 2: v projection (128 -> 128 ch), f16 [B][C][pos]
// ---------------------------------------------------------------------------
__global__ void __launch_bounds__(256)
proj_v_kernel(const float* __restrict__ xh,
              const float* __restrict__ Wv, const float* __restrict__ bv,
              _Float16* __restrict__ vh)
{
    __shared__ float xs[VC][65];
    const int tx = threadIdx.x;
    const int b  = blockIdx.x / (NPOS / 64);
    const int it = blockIdx.x % (NPOS / 64);
    const int i0 = it * 64;

    for (int idx = tx; idx < VC * 64; idx += 256) {
        int cc = idx >> 6, il = idx & 63;
        xs[cc][il] = xh[((size_t)b * VC + cc) * NPOS + i0 + il];
    }
    __syncthreads();

    const int il = tx & 63;
    const int c0 = (tx >> 6) * 32;

    float acc[32];
#pragma unroll
    for (int k = 0; k < 32; ++k) acc[k] = bv[c0 + k];

    for (int cc = 0; cc < VC; ++cc) {
        float xv = xs[cc][il];
#pragma unroll
        for (int k = 0; k < 32; ++k)
            acc[k] = fmaf(Wv[(size_t)(c0 + k) * VC + cc], xv, acc[k]);
    }
#pragma unroll
    for (int k = 0; k < 32; ++k)
        vh[((size_t)b * VC + c0 + k) * NPOS + i0 + il] = (_Float16)acc[k];
}

// ---------------------------------------------------------------------------
// A/B fragment loader: elements 0..7 at +0, 8..15 at +16 halves; caller adds
// the half-wave K offset (0/8 halves). Two b128 loads (global or ds).
// ---------------------------------------------------------------------------
__device__ __forceinline__ v16h load_frag16(const _Float16* base) {
    const v8h* p = (const v8h*)base;
    v8h lo = p[0];
    v8h hi = p[2];
    v16h r;
#pragma unroll
    for (int e = 0; e < 8; ++e) { r[e] = lo[e]; r[8 + e] = hi[e]; }
    return r;
}

// cooperative async stage of one 64-key chunk: K (4KB) + V (16KB) -> LDS
__device__ __forceinline__ void stage_chunk(int tid, const _Float16* kg,
                                            const _Float16* vg, int j0,
                                            _Float16* kdst, _Float16* vdst)
{
    const _Float16* ks = kg + (size_t)j0 * 32;
#pragma unroll
    for (int it = 0; it < 2; ++it) {                 // 256 x 16B for K
        int m = tid + it * 128;
        async_ld16(ks + m * 8, kdst + m * 8);
    }
#pragma unroll
    for (int it = 0; it < 8; ++it) {                 // 1024 x 16B for V
        int m = tid + it * 128;
        int c = m >> 3, k8 = m & 7;
        async_ld16(vg + (size_t)c * NPOS + j0 + k8 * 8, vdst + c * 64 + k8 * 8);
    }
}
#define ASYNC_PER_STAGE 10

// ---------------------------------------------------------------------------
// Kernel 3: flash attention. 4 waves/WG, 64 queries/WG, double-buffered
// async K/V staging, 64-key chunks, row-sum via a WMMA "ones" column.
// ---------------------------------------------------------------------------
__global__ void __launch_bounds__(128)
attn_kernel(const _Float16* __restrict__ qh, const _Float16* __restrict__ kh,
            const _Float16* __restrict__ vh, const float* __restrict__ xh,
            const float* __restrict__ gamma, float* __restrict__ out)
{
    __shared__ _Float16 kbuf[2][KCH * 32];        // 2 x 4 KB
    __shared__ _Float16 vbuf[2][VC * KCH];        // 2 x 16 KB
    __shared__ _Float16 Pbuf[4][16 * KCH];        // 4 x 2 KB
    __shared__ float    obuf[4][16][17];          // epilogue transpose

    const int tid  = threadIdx.x;
    const int lane = tid & 31;
    const int wv   = tid >> 5;
    const int b    = blockIdx.x / (NPOS / 64);
    const int it   = blockIdx.x % (NPOS / 64);
    const int i0w  = it * 64 + wv * 16;           // this wave's 16 queries
    const int ln   = lane & 15;
    const int koff = (lane < 16) ? 0 : 8;         // half-wave K offset (halves)
    const int mhalf= (lane < 16) ? 0 : 8;         // half-wave row offset (C/D)

    const _Float16* kbase = kh + (size_t)b * NPOS * 32;
    const _Float16* vbase = vh + (size_t)b * VC * NPOS;

    // Q A-fragment, loaded once
    const v16h qa = load_frag16(qh + ((size_t)b * NPOS + i0w) * 32
                                   + (size_t)ln * 32 + koff);

    // constant "ones column" B-fragment: column 0 (lanes ln==0) is all 1.0
    v16h onesb;
#pragma unroll
    for (int e = 0; e < 16; ++e) onesb[e] = (ln == 0) ? (_Float16)1.f : (_Float16)0.f;

    float m[8];
    v8f   O[9];                                    // 8 channel tiles + row-sum tile
#pragma unroll
    for (int r = 0; r < 8; ++r) m[r] = -3.0e38f;
#pragma unroll
    for (int t = 0; t < 9; ++t)
#pragma unroll
        for (int e = 0; e < 8; ++e) O[t][e] = 0.f;

    stage_chunk(tid, kbase, vbase, 0, kbuf[0], vbuf[0]);

    for (int ch = 0; ch < NCH; ++ch) {
        const int cur = ch & 1;
        if (ch + 1 < NCH) {
            stage_chunk(tid, kbase, vbase, (ch + 1) * KCH,
                        kbuf[cur ^ 1], vbuf[cur ^ 1]);
            WAIT_ASYNC(ASYNC_PER_STAGE);           // chunk `ch` resident,
        } else {                                   // next chunk still in flight
            WAIT_ASYNC(0);
        }
        __syncthreads();

        const _Float16* kb = kbuf[cur];
        const _Float16* vb = vbuf[cur];

        // S = q . k : batch all 4 fragment loads, then 4 back-to-back WMMAs
        v16h kf[4];
#pragma unroll
        for (int tt = 0; tt < 4; ++tt)
            kf[tt] = load_frag16(kb + (size_t)(tt * 16 + ln) * 32 + koff);

        v8f S[4];
        v8f z;
#pragma unroll
        for (int e = 0; e < 8; ++e) z[e] = 0.f;
#pragma unroll
        for (int tt = 0; tt < 4; ++tt)
            S[tt] = __builtin_amdgcn_wmma_f32_16x16x32_f16(false, qa, false, kf[tt],
                                                           (short)0, z, false, false);

        // online softmax (max only; row-sum comes from the ones-column WMMA)
        float sc[8];
#pragma unroll
        for (int r = 0; r < 8; ++r) {
            float mx = fmaxf(fmaxf(S[0][r], S[1][r]), fmaxf(S[2][r], S[3][r]));
            mx = xor16_max(mx);
            float mn = fmaxf(m[r], mx);
            sc[r] = __expf(m[r] - mn);
            m[r]  = mn;
#pragma unroll
            for (int tt = 0; tt < 4; ++tt) S[tt][r] = __expf(S[tt][r] - mn);
        }
#pragma unroll
        for (int t = 0; t < 9; ++t)
#pragma unroll
            for (int r = 0; r < 8; ++r) O[t][r] *= sc[r];

        // P (C/D layout) -> per-wave LDS [16 x 64] f16
        _Float16* Pw = Pbuf[wv];
#pragma unroll
        for (int tt = 0; tt < 4; ++tt)
#pragma unroll
            for (int r = 0; r < 8; ++r)
                Pw[(mhalf + r) * KCH + tt * 16 + ln] = (_Float16)S[tt][r];
        LDS_FENCE();

        // reload P as A fragments (K = keys 0..31 / 32..63)
        const v16h pa0 = load_frag16(Pw + (size_t)ln * KCH + koff);
        const v16h pa1 = load_frag16(Pw + (size_t)ln * KCH + 32 + koff);

        // O += P x V^T : software-pipelined fragment loads one tile ahead
        v16h vfa0 = load_frag16(vb + (size_t)ln * KCH + koff);
        v16h vfa1 = load_frag16(vb + (size_t)ln * KCH + 32 + koff);
#pragma unroll
        for (int ct = 0; ct < 8; ++ct) {
            v16h vfb0, vfb1;
            if (ct < 7) {
                vfb0 = load_frag16(vb + (size_t)((ct + 1) * 16 + ln) * KCH + koff);
                vfb1 = load_frag16(vb + (size_t)((ct + 1) * 16 + ln) * KCH + 32 + koff);
            }
            O[ct] = __builtin_amdgcn_wmma_f32_16x16x32_f16(false, pa0, false, vfa0,
                                                           (short)0, O[ct], false, false);
            O[ct] = __builtin_amdgcn_wmma_f32_16x16x32_f16(false, pa1, false, vfa1,
                                                           (short)0, O[ct], false, false);
            vfa0 = vfb0;
            vfa1 = vfb1;
        }
        // row-sum tile
        O[8] = __builtin_amdgcn_wmma_f32_16x16x32_f16(false, pa0, false, onesb,
                                                      (short)0, O[8], false, false);
        O[8] = __builtin_amdgcn_wmma_f32_16x16x32_f16(false, pa1, false, onesb,
                                                      (short)0, O[8], false, false);

        __syncthreads();   // protect staging buffers before next overwrite
    }

    // epilogue: l from O[8] column 0 (lanes 0 / 16), broadcast to the half
    const float g = gamma[0];
    float inv[8];
#pragma unroll
    for (int r = 0; r < 8; ++r) {
        float lr = __shfl(O[8][r], (lane < 16) ? 0 : 16, 32);
        inv[r] = g / lr;
    }

    // transpose each 16x16 tile through LDS for coalesced residual+store
#pragma unroll
    for (int t = 0; t < 8; ++t) {
#pragma unroll
        for (int r = 0; r < 8; ++r)
            obuf[wv][mhalf + r][ln] = O[t][r] * inv[r];
        LDS_FENCE();
        const int ir = lane & 15, cg = lane >> 4;
#pragma unroll
        for (int cc = 0; cc < 8; ++cc) {
            int c = t * 16 + cg * 8 + cc;
            size_t idx = ((size_t)b * VC + c) * NPOS + i0w + ir;
            out[idx] = obuf[wv][ir][cg * 8 + cc] + xh[idx];
        }
        LDS_FENCE();
    }
}

extern "C" void kernel_launch(void* const* d_in, const int* in_sizes, int n_in,
                              void* d_out, int out_size, void* d_ws, size_t ws_size,
                              hipStream_t stream) {
    (void)in_sizes; (void)n_in; (void)out_size; (void)ws_size;
    const float* x     = (const float*)d_in[0];
    const float* xh    = (const float*)d_in[1];
    const float* Wq    = (const float*)d_in[2];
    const float* bq    = (const float*)d_in[3];
    const float* Wk    = (const float*)d_in[4];
    const float* bk    = (const float*)d_in[5];
    const float* Wv    = (const float*)d_in[6];
    const float* bv    = (const float*)d_in[7];
    const float* gamma = (const float*)d_in[8];
    float* out = (float*)d_out;

    char* w = (char*)d_ws;
    _Float16* qh = (_Float16*)(w);                 // 1 MB: [B][4096][32]
    _Float16* kh = (_Float16*)(w + (1u << 20));    // 1 MB: [B][4096][32]
    _Float16* vh = (_Float16*)(w + (2u << 20));    // 4 MB: [B][128][4096]

    proj_qk_kernel<<<BATCH * (NPOS / 64), 64, 0, stream>>>(x, Wq, bq, Wk, bk, qh, kh);
    proj_v_kernel <<<BATCH * (NPOS / 64), 256, 0, stream>>>(xh, Wv, bv, vh);
    attn_kernel   <<<BATCH * (NPOS / 64), 128, 0, stream>>>(qh, kh, vh, xh, gamma, out);
}
